// CR_86955907875291
// MI455X (gfx1250) — matrix operand order repaired and checked by reading, not compiled, not run on hardware
//
#include <hip/hip_runtime.h>
#include <hip/hip_bf16.h>

typedef __attribute__((ext_vector_type(16))) __bf16 v16bf;
typedef __attribute__((ext_vector_type(8)))  float  v8f;

#define C    256
#define MBLK 64

#if __has_builtin(__builtin_amdgcn_sched_barrier)
#define SCHED_FENCE() __builtin_amdgcn_sched_barrier(0)
#else
#define SCHED_FENCE()
#endif

// round-to-nearest-even f32 -> bf16 bits (scalar path, used in epilogues)
__device__ __forceinline__ unsigned short f2bf(float f) {
    unsigned u = __float_as_uint(f);
    return (unsigned short)((u + 0x7FFFu + ((u >> 16) & 1u)) >> 16);
}

// packed f32x2 -> bf16x2 bits via hip_bf16 (lowers to packed cvt on gfx1250)
__device__ __forceinline__ unsigned f2bf_pk(float lo, float hi) {
    union { __hip_bfloat162 h; unsigned u; } cv;
    cv.h = __float22bfloat162_rn(make_float2(lo, hi));
    return cv.u;
}

union FragBF { v16bf v; uint4 q[2]; };

// --- kernel 1: LSQ quantize both weights to bf16 integer grid q = round(clip(w/a,-8,7)) ---
__global__ void CR_lsq_quant(const float* __restrict__ w1, const float* __restrict__ a1p,
                             const float* __restrict__ w2, const float* __restrict__ a2p,
                             unsigned short* __restrict__ q1, unsigned short* __restrict__ q2) {
    const int i = blockIdx.x * blockDim.x + threadIdx.x;
    const float inv1 = 1.0f / *a1p;
    const float inv2 = 1.0f / *a2p;
    const float v1 = rintf(fminf(fmaxf(w1[i] * inv1, -8.0f), 7.0f));
    const float v2 = rintf(fminf(fmaxf(w2[i] * inv2, -8.0f), 7.0f));
    // small integers: bf16 truncation of f32 bits is exact
    q1[i] = (unsigned short)(__float_as_uint(v1) >> 16);
    q2[i] = (unsigned short)(__float_as_uint(v2) >> 16);
}

// --- kernel 2: fused  out = sigmoid(a2*(relu(a1*(x@q1^T)+b1)@q2^T)+b2) ---
__global__ __launch_bounds__(256)
void CR_fused_mlp(const float* __restrict__ x,
                  const unsigned short* __restrict__ q1,
                  const float* __restrict__ b1, const float* __restrict__ a1p,
                  const unsigned short* __restrict__ q2,
                  const float* __restrict__ b2, const float* __restrict__ a2p,
                  float* __restrict__ out) {
    __shared__ __align__(16) unsigned short xs[MBLK * C]; // 32 KB bf16 x-tile
    __shared__ __align__(16) unsigned short hs[MBLK * C]; // 32 KB bf16 h-tile

    const int  tid  = threadIdx.x;
    const int  lane = tid & 31;
    const int  wave = tid >> 5;
    const int  m16  = lane & 15;   // M (A) / N (B,C) index within tile
    const int  half = lane >> 4;   // lane group: selects K-half (A,B) / M-half (C)
    const long gRow0 = (long)blockIdx.x * MBLK;
    const float alpha1 = *a1p;
    const float alpha2 = *a2p;

    // wave -> (one 16-row tile, 8 column tiles): 8 waves = 4 row tiles x 2 col halves
    const int rowTile = (wave >> 1) * 16;       // 0,16,32,48
    const int colBase = (wave & 1) * 128;       // 0 or 128

    // ---- stage x tile: global f32 -> LDS bf16 (each thread: 64 floats of one row) ----
    {
        const int row = tid >> 2;          // 0..63
        const int qd  = (tid & 3) * 64;    // column start within row
        const float4* src = (const float4*)(x + (gRow0 + row) * C + qd);
        uint4* dst = (uint4*)(xs + row * C + qd);
        #pragma unroll
        for (int i = 0; i < 8; ++i) {
            float4 f0 = src[2 * i];
            float4 f1 = src[2 * i + 1];
            uint4 p;
            p.x = f2bf_pk(f0.x, f0.y);
            p.y = f2bf_pk(f0.z, f0.w);
            p.z = f2bf_pk(f1.x, f1.y);
            p.w = f2bf_pk(f1.z, f1.w);
            dst[i] = p;
        }
    }
    __syncthreads();

    // ---- layer 1: h = relu(alpha1 * (x @ q1^T) + b1), kept in LDS as bf16 ----
    {
        float bias[8];
        #pragma unroll
        for (int j = 0; j < 8; ++j) bias[j] = b1[colBase + j * 16 + m16];

        v8f acc[8] = {};
        // A (16x32 bf16): lanes 0-15 hold K={0-7,16-23}, lanes 16-31 K={8-15,24-31}
        const unsigned short* arow = xs + (rowTile + m16) * C + half * 8;
        // B (32x16 bf16): lanes 0-15 hold K={0-7,16-23}, lanes 16-31 K={8-15,24-31}
        const unsigned short* brow = q1 + (colBase + m16) * C + half * 8;
        #pragma unroll 1
        for (int kc = 0; kc < 8; ++kc) {
            FragBF A;
            A.q[0] = *(const uint4*)(arow + kc * 32);
            A.q[1] = *(const uint4*)(arow + kc * 32 + 16);
            FragBF Bf[8];
            #pragma unroll
            for (int j = 0; j < 8; ++j) {
                const unsigned short* bp = brow + j * 16 * C + kc * 32;
                Bf[j].q[0] = *(const uint4*)(bp);
                Bf[j].q[1] = *(const uint4*)(bp + 16);
            }
            SCHED_FENCE();   // keep all 16 B loads + 2 A loads issued before the WMMAs
            #pragma unroll
            for (int j = 0; j < 8; ++j)
                acc[j] = __builtin_amdgcn_wmma_f32_16x16x32_bf16(
                             false, A.v, false, Bf[j].v, (short)0, acc[j], false, false);
            SCHED_FENCE();
        }
        #pragma unroll
        for (int j = 0; j < 8; ++j) {
            const int coln = colBase + j * 16 + m16;
            #pragma unroll
            for (int v = 0; v < 8; ++v) {
                const int row = rowTile + (half ? v + 8 : v);
                float val = fmaxf(acc[j][v] * alpha1 + bias[j], 0.0f);
                hs[row * C + coln] = f2bf(val);
            }
        }
    }
    __syncthreads();

    // ---- layer 2: out = sigmoid(alpha2 * (h @ q2^T) + b2) ----
    {
        float bias[8];
        #pragma unroll
        for (int j = 0; j < 8; ++j) bias[j] = b2[colBase + j * 16 + m16];

        v8f acc[8] = {};
        const unsigned short* arow = hs + (rowTile + m16) * C + half * 8;
        const unsigned short* brow = q2 + (colBase + m16) * C + half * 8;
        #pragma unroll 1
        for (int kc = 0; kc < 8; ++kc) {
            FragBF A;
            A.q[0] = *(const uint4*)(arow + kc * 32);
            A.q[1] = *(const uint4*)(arow + kc * 32 + 16);
            FragBF Bf[8];
            #pragma unroll
            for (int j = 0; j < 8; ++j) {
                const unsigned short* bp = brow + j * 16 * C + kc * 32;
                Bf[j].q[0] = *(const uint4*)(bp);
                Bf[j].q[1] = *(const uint4*)(bp + 16);
            }
            SCHED_FENCE();
            #pragma unroll
            for (int j = 0; j < 8; ++j)
                acc[j] = __builtin_amdgcn_wmma_f32_16x16x32_bf16(
                             false, A.v, false, Bf[j].v, (short)0, acc[j], false, false);
            SCHED_FENCE();
        }
        #pragma unroll
        for (int j = 0; j < 8; ++j) {
            const int coln = colBase + j * 16 + m16;
            #pragma unroll
            for (int v = 0; v < 8; ++v) {
                const int row = rowTile + (half ? v + 8 : v);
                float val = acc[j][v] * alpha2 + bias[j];
                val = 1.0f / (1.0f + __expf(-val));
                out[(gRow0 + row) * C + coln] = val;
            }
        }
    }
}

extern "C" void kernel_launch(void* const* d_in, const int* in_sizes, int n_in,
                              void* d_out, int out_size, void* d_ws, size_t ws_size,
                              hipStream_t stream) {
    // setup_inputs order: x, w1, b1, alpha1, w2, b2, alpha2
    const float* x  = (const float*)d_in[0];
    const float* w1 = (const float*)d_in[1];
    const float* b1 = (const float*)d_in[2];
    const float* a1 = (const float*)d_in[3];
    const float* w2 = (const float*)d_in[4];
    const float* b2 = (const float*)d_in[5];
    const float* a2 = (const float*)d_in[6];
    float* out = (float*)d_out;

    // workspace: two 256x256 bf16 quantized-integer weight grids (128 KB each)
    unsigned short* q1 = (unsigned short*)d_ws;
    unsigned short* q2 = q1 + C * C;

    CR_lsq_quant<<<(C * C) / 256, 256, 0, stream>>>(w1, a1, w2, a2, q1, q2);

    const int N = in_sizes[0] / C;  // 262144 rows
    CR_fused_mlp<<<N / MBLK, 256, 0, stream>>>(x, q1, b1, a1, q2, b2, a2, out);
}
